// Attention_46686294507947
// MI455X (gfx1250) — compile-verified, matrix operand order checked by reading, not run on hardware
//
#include <hip/hip_runtime.h>
#include <hip/hip_bf16.h>
#include <cstddef>
#include <cstdint>

// ---------------------------------------------------------------------------
// MHA block for MI455X (gfx1250, wave32, WMMA + TDM).
//   0) x -> f16; w_qkv, w_out -> f16 transposed (one-time layout/precision)
//   1) qkv[f16]  = xh @ wqkvT'   (128x128x32, TDM-staged double-buffered LDS)
//   2) attn[f16] = flash attention (K tile via TDM, V transposed by threads)
//   3) out[f32]  = attn @ woutT' + bias
// GEMM/attention staging of f16 tiles is pure byte movement -> use the
// Tensor Data Mover (TENSOR_LOAD_TO_LDS) with LDS row padding, tracked by
// TENSORcnt; matrix math in v_wmma_f32_16x16x32_f16 with f32 accumulate.
// ---------------------------------------------------------------------------

typedef __attribute__((ext_vector_type(16))) _Float16 v16h;
typedef __attribute__((ext_vector_type(8)))  _Float16 v8h;
typedef __attribute__((ext_vector_type(4)))  _Float16 v4h;
typedef __attribute__((ext_vector_type(8)))  float    v8f;

typedef __attribute__((ext_vector_type(4))) unsigned int tdm_g0_t;
typedef __attribute__((ext_vector_type(8))) int          tdm_g1_t;
typedef __attribute__((ext_vector_type(4))) int          tdm_g2_t;

// ---- WMMA fragment helpers (ISA 7.12.2 layouts, wave32) --------------------
__device__ __forceinline__ v16h load_frag16(const _Float16* __restrict__ p,
                                            int ld, int lane) {
  const int m  = lane & 15;
  const int kb = (lane >> 4) << 3;
  v16h f;
#pragma unroll
  for (int j = 0; j < 8; ++j) {
    const int k = kb + ((j < 4) ? (2 * j) : (16 + 2 * (j - 4)));
    f[2 * j]     = p[m * ld + k];
    f[2 * j + 1] = p[m * ld + k + 1];
  }
  return f;
}

__device__ __forceinline__ v8f wmma_f32_f16(v16h a, v16h b, v8f c) {
  return __builtin_amdgcn_wmma_f32_16x16x32_f16(false, a, false, b,
                                                (short)0, c, false, false);
}

// ---- Tensor Data Mover: 2D f16 tile (tile_k x tile_rows) -> LDS ------------
// D# per ISA 8.3/8.4. global addr = tile start; y rows advance by
// stride_elems (data_size units). LDS rows padded via pad_interval/amount:
//   pad_interval: after (1<<pi) qwords stored   (table: 3 -> 16 DW = 64B row,
//                                                       4 -> 32 DW = 128B row)
//   pad_amount:   insert (pa+1) DWORDs           (3 -> 16B pad)
__device__ __forceinline__ void tdm_load_tile_f16(
    unsigned lds_byte_off, const _Float16* __restrict__ gsrc,
    unsigned tile_k, unsigned tile_rows, unsigned stride_elems,
    unsigned pad_interval, unsigned pad_amount) {
  const unsigned long long ga = (unsigned long long)(uintptr_t)gsrc;
  tdm_g0_t g0;
  g0[0] = 1u;                                    // count=1, user mode
  g0[1] = lds_byte_off;                          // lds_addr (bytes)
  g0[2] = (unsigned)ga;                          // global_addr[31:0]
  g0[3] = (unsigned)((ga >> 32) & 0x01ffffffu)   // global_addr[56:32]
          | (2u << 30);                          // type = 2 ("image")
  tdm_g1_t g1;
  g1[0] = (int)((1u << 16)                       // data_size = 2 bytes
                | (1u << 20)                     // pad_enable
                | (pad_interval << 22)
                | (pad_amount << 25));
  g1[1] = (int)(0x4000u << 16);                  // tensor_dim0[15:0] (no clip)
  g1[2] = (int)(0x4000u << 16);                  // dim0 hi=0 | tensor_dim1 lo
  g1[3] = (int)(tile_k << 16);                   // dim1 hi=0 | tile_dim0
  g1[4] = (int)(tile_rows & 0xffffu);            // tile_dim1 (tile_dim2=0)
  g1[5] = (int)stride_elems;                     // tensor_dim0_stride[31:0]
  g1[6] = 0;                                     // stride hi | dim1_stride lo
  g1[7] = 0;
  const tdm_g2_t zz = (tdm_g2_t)0;
#if defined(__clang_major__) && (__clang_major__ >= 23)
  const tdm_g1_t z8 = (tdm_g1_t)0;
  __builtin_amdgcn_tensor_load_to_lds(g0, g1, zz, zz, z8, 0);
#else
  __builtin_amdgcn_tensor_load_to_lds(g0, g1, zz, zz, 0);
#endif
}

__device__ __forceinline__ unsigned lds_off32(const void* p) {
  // generic LDS pointer: addr[31:0] is the LDS byte offset (ISA 10.2)
  return (unsigned)(size_t)p;
}

// ---------------------------------------------------------------------------
// Elementwise f32 -> f16 (n % 1024 == 0)
// ---------------------------------------------------------------------------
__global__ __launch_bounds__(256)
void cvt_f16(const float* __restrict__ X, _Float16* __restrict__ Y) {
  const size_t i = ((size_t)blockIdx.x * 256 + threadIdx.x) * 4;
  const float4 v = *(const float4*)(X + i);
  v4h o;
  o[0] = (_Float16)v.x; o[1] = (_Float16)v.y;
  o[2] = (_Float16)v.z; o[3] = (_Float16)v.w;
  *(v4h*)(Y + i) = o;
}

// ---------------------------------------------------------------------------
// Transpose + convert: W[K,N] f32 -> Wt[N,K] f16.  Grid (N/32, K/32).
// ---------------------------------------------------------------------------
__global__ __launch_bounds__(256)
void transpose_cvt_f16(const float* __restrict__ W, _Float16* __restrict__ Wt,
                       int K, int N) {
  __shared__ _Float16 t[32][33];
  const int k0 = blockIdx.y * 32, n0 = blockIdx.x * 32;
  const int tn = threadIdx.x & 31, tr = threadIdx.x >> 5;
#pragma unroll
  for (int p = 0; p < 4; ++p)
    t[tr + p * 8][tn] = (_Float16)W[(size_t)(k0 + tr + p * 8) * N + n0 + tn];
  __syncthreads();
#pragma unroll
  for (int p = 0; p < 4; ++p)
    Wt[(size_t)(n0 + tr + p * 8) * K + k0 + tn] = t[tn][tr + p * 8];
}

// ---------------------------------------------------------------------------
// GEMM: C[M,N] = A[M,K] @ Bt[N,K]^T (+bias). A, Bt f16 row-major.
// 128x128x32 tiles, 8 waves in 4x2, 8 WMMA per wave per K-step.
// TDM stages both tiles (wave 0 issues, TENSORcnt-gated), double-buffered.
// ---------------------------------------------------------------------------
template <bool HASBIAS, bool OUTF16>
__global__ __launch_bounds__(256)
void gemm_wmma(const _Float16* __restrict__ A, const _Float16* __restrict__ Bt,
               void* __restrict__ Cv, const float* __restrict__ bias,
               int M, int N, int K) {
  constexpr int BM = 128, BN = 128, BK = 32, LD = BK + 8;  // 80B LDS rows
  __shared__ _Float16 As[2][BM][LD];
  __shared__ _Float16 Bs[2][BN][LD];

  const int tid  = threadIdx.x;
  const int lane = tid & 31;
  const int wave = tid >> 5;
  const int m0 = blockIdx.y * BM, n0 = blockIdx.x * BN;
  const int wm = (wave >> 1) * 32;
  const int wn = (wave & 1) * 64;

  const _Float16* atile = A  + (size_t)m0 * K;
  const _Float16* btile = Bt + (size_t)n0 * K;
  const unsigned asbase = lds_off32(&As[0][0][0]);
  const unsigned bsbase = lds_off32(&Bs[0][0][0]);
  constexpr unsigned BUFB = (unsigned)BM * LD * 2;  // bytes per buffer

  v8f acc[2][4];
#pragma unroll
  for (int i = 0; i < 2; ++i)
#pragma unroll
    for (int j = 0; j < 4; ++j) acc[i][j] = (v8f){};

  if (wave == 0) {
    tdm_load_tile_f16(asbase, atile, BK, BM, (unsigned)K, 3u, 3u);
    tdm_load_tile_f16(bsbase, btile, BK, BN, (unsigned)K, 3u, 3u);
    __builtin_amdgcn_s_wait_tensorcnt(0);
  }
  __syncthreads();

  const int KT = K / BK;
  for (int kt = 0; kt < KT; ++kt) {
    const int cur = kt & 1;
    if (wave == 0 && kt + 1 < KT) {
      const unsigned nb = (unsigned)(cur ^ 1) * BUFB;
      tdm_load_tile_f16(asbase + nb, atile + (kt + 1) * BK, BK, BM,
                        (unsigned)K, 3u, 3u);
      tdm_load_tile_f16(bsbase + nb, btile + (kt + 1) * BK, BK, BN,
                        (unsigned)K, 3u, 3u);
    }

    v16h af[2], bf[4];
#pragma unroll
    for (int i = 0; i < 2; ++i)
      af[i] = load_frag16(&As[cur][wm + i * 16][0], LD, lane);
#pragma unroll
    for (int j = 0; j < 4; ++j)
      bf[j] = load_frag16(&Bs[cur][wn + j * 16][0], LD, lane);
#pragma unroll
    for (int i = 0; i < 2; ++i)
#pragma unroll
      for (int j = 0; j < 4; ++j)
        acc[i][j] = wmma_f32_f16(af[i], bf[j], acc[i][j]);

    if (wave == 0 && kt + 1 < KT) __builtin_amdgcn_s_wait_tensorcnt(0);
    __syncthreads();
  }

  const int r8 = (lane >> 4) << 3;
  const int cn = lane & 15;
  float bv[4];
#pragma unroll
  for (int j = 0; j < 4; ++j)
    bv[j] = HASBIAS ? bias[n0 + wn + j * 16 + cn] : 0.0f;

#pragma unroll
  for (int i = 0; i < 2; ++i)
#pragma unroll
    for (int j = 0; j < 4; ++j) {
      const int colg = n0 + wn + j * 16 + cn;
#pragma unroll
      for (int r = 0; r < 8; ++r) {
        const int rowg = m0 + wm + i * 16 + r + r8;
        if (OUTF16)
          ((_Float16*)Cv)[(size_t)rowg * N + colg] = (_Float16)acc[i][j][r];
        else
          ((float*)Cv)[(size_t)rowg * N + colg] = acc[i][j][r] + bv[j];
      }
    }
}

// ---------------------------------------------------------------------------
// Flash attention (f16 in/out). qkv: [b=2,n=2048,3*1024] f16 (q|k|v).
// out: [b*n, 1024] f16. Grid (n/128, heads, b), 8 waves x 16 query rows.
// K tile staged by TDM; V tile transposed by threads; softmax in f32.
// ---------------------------------------------------------------------------
__global__ __launch_bounds__(256)
void flash_attn_wmma(const _Float16* __restrict__ qkv,
                     _Float16* __restrict__ out) {
  constexpr int NSEQ = 2048, D = 64, INNER = 1024, QLD = 3 * INNER;
  constexpr int LDK = D + 8;  // 144B LDS rows
  __shared__ _Float16 Ks[64][LDK];
  __shared__ _Float16 Vs[64][LDK];     // [d][key]
  __shared__ _Float16 Ps[8][16][LDK];  // per-wave P tile

  const int tid  = threadIdx.x;
  const int lane = tid & 31;
  const int wave = tid >> 5;
  const int h  = blockIdx.y;
  const int bi = blockIdx.z;
  const int qbase = blockIdx.x * 128 + wave * 16;
  const int m  = lane & 15;
  const int hi = lane >> 4;
  const unsigned ksbase = lds_off32(&Ks[0][0]);

  v16h qa0, qa1;
  {
    const _Float16* qrow =
        qkv + ((size_t)(bi * NSEQ + qbase + m)) * QLD + h * D;
    const _Float16 sc = (_Float16)0.125f;  // 1/sqrt(64), exact in f16
    const int kb = hi << 3;
#pragma unroll
    for (int j = 0; j < 8; ++j) {
      const int k = kb + ((j < 4) ? (2 * j) : (16 + 2 * (j - 4)));
      qa0[2 * j]     = qrow[k] * sc;
      qa0[2 * j + 1] = qrow[k + 1] * sc;
      qa1[2 * j]     = qrow[32 + k] * sc;
      qa1[2 * j + 1] = qrow[32 + k + 1] * sc;
    }
  }

  float mrun[8], lrun[8];
  v8f o[4];
#pragma unroll
  for (int r = 0; r < 8; ++r) { mrun[r] = -3.0e38f; lrun[r] = 0.0f; }
#pragma unroll
  for (int t = 0; t < 4; ++t) o[t] = (v8f){};

  const int krow  = tid >> 2;
  const int dbase = (tid & 3) * 16;

  for (int kv0 = 0; kv0 < NSEQ; kv0 += 64) {
    __syncthreads();
    // K tile [64 keys x 64 d] via TDM (row 128B -> pad_interval 4, +16B pad)
    if (wave == 0)
      tdm_load_tile_f16(ksbase,
                        qkv + (size_t)(bi * NSEQ + kv0) * QLD + INNER + h * D,
                        D, 64, (unsigned)QLD, 4u, 3u);
    // V tile transposed by all threads
    const _Float16* vg = qkv + (size_t)(bi * NSEQ + kv0 + krow) * QLD
                         + 2 * INNER + h * D + dbase;
    const v8h v0 = *(const v8h*)vg;
    const v8h v1 = *(const v8h*)(vg + 8);
#pragma unroll
    for (int i = 0; i < 8; ++i) {
      Vs[dbase + i][krow]     = v0[i];
      Vs[dbase + 8 + i][krow] = v1[i];
    }
    if (wave == 0) __builtin_amdgcn_s_wait_tensorcnt(0);
    __syncthreads();

    v8f s[4];
#pragma unroll
    for (int t = 0; t < 4; ++t) {
      const v16h b0 = load_frag16(&Ks[t * 16][0], LDK, lane);
      const v16h b1 = load_frag16(&Ks[t * 16][32], LDK, lane);
      v8f z = {};
      z = wmma_f32_f16(qa0, b0, z);
      z = wmma_f32_f16(qa1, b1, z);
      s[t] = z;
    }

#pragma unroll
    for (int r = 0; r < 8; ++r) {
      float mx = s[0][r];
#pragma unroll
      for (int t = 1; t < 4; ++t) mx = fmaxf(mx, s[t][r]);
      mx = fmaxf(mx, __shfl_xor(mx, 1, 32));
      mx = fmaxf(mx, __shfl_xor(mx, 2, 32));
      mx = fmaxf(mx, __shfl_xor(mx, 4, 32));
      mx = fmaxf(mx, __shfl_xor(mx, 8, 32));
      const float mn   = fmaxf(mrun[r], mx);
      const float corr = __expf(mrun[r] - mn);
      float rs = 0.0f;
#pragma unroll
      for (int t = 0; t < 4; ++t) {
        const float p = __expf(s[t][r] - mn);
        rs += p;
        Ps[wave][r + hi * 8][t * 16 + m] = (_Float16)p;
      }
      rs += __shfl_xor(rs, 1, 32);
      rs += __shfl_xor(rs, 2, 32);
      rs += __shfl_xor(rs, 4, 32);
      rs += __shfl_xor(rs, 8, 32);
      lrun[r] = lrun[r] * corr + rs;
      mrun[r] = mn;
#pragma unroll
      for (int t = 0; t < 4; ++t) o[t][r] *= corr;
    }

    const v16h pa0 = load_frag16(&Ps[wave][0][0], LDK, lane);
    const v16h pa1 = load_frag16(&Ps[wave][0][32], LDK, lane);
#pragma unroll
    for (int t = 0; t < 4; ++t) {
      const v16h vb0 = load_frag16(&Vs[t * 16][0], LDK, lane);
      const v16h vb1 = load_frag16(&Vs[t * 16][32], LDK, lane);
      o[t] = wmma_f32_f16(pa0, vb0, o[t]);
      o[t] = wmma_f32_f16(pa1, vb1, o[t]);
    }
  }

#pragma unroll
  for (int r = 0; r < 8; ++r) {
    const float inv = 1.0f / lrun[r];
    const int row = qbase + r + hi * 8;
    _Float16* og = out + ((size_t)(bi * NSEQ + row)) * INNER + h * D;
#pragma unroll
    for (int t = 0; t < 4; ++t)
      og[t * 16 + m] = (_Float16)(o[t][r] * inv);
  }
}

// ---------------------------------------------------------------------------
extern "C" void kernel_launch(void* const* d_in, const int* in_sizes, int n_in,
                              void* d_out, int out_size, void* d_ws,
                              size_t ws_size, hipStream_t stream) {
  (void)in_sizes; (void)n_in; (void)out_size; (void)ws_size;
  const float* x     = (const float*)d_in[0];
  const float* w_qkv = (const float*)d_in[1];
  const float* w_out = (const float*)d_in[2];
  const float* b_out = (const float*)d_in[3];
  float* out = (float*)d_out;

  constexpr int ROWS = 2 * 2048;
  constexpr int DIM = 1024, INNER = 1024, QKVN = 3 * INNER;

  _Float16* xh    = (_Float16*)d_ws;
  _Float16* wqkvT = xh    + (size_t)ROWS * DIM;
  _Float16* woutT = wqkvT + (size_t)QKVN * DIM;
  _Float16* qkvh  = woutT + (size_t)DIM * INNER;
  _Float16* attnh = qkvh  + (size_t)ROWS * QKVN;

  cvt_f16<<<(ROWS * DIM) / (256 * 4), 256, 0, stream>>>(x, xh);
  transpose_cvt_f16<<<dim3(QKVN / 32, DIM / 32), 256, 0, stream>>>(
      w_qkv, wqkvT, DIM, QKVN);
  transpose_cvt_f16<<<dim3(INNER / 32, DIM / 32), 256, 0, stream>>>(
      w_out, woutT, INNER, DIM);

  gemm_wmma<false, true><<<dim3(QKVN / 128, ROWS / 128), 256, 0, stream>>>(
      xh, wqkvT, qkvh, nullptr, ROWS, QKVN, DIM);

  flash_attn_wmma<<<dim3(2048 / 128, 16, 2), 256, 0, stream>>>(qkvh, attnh);

  gemm_wmma<true, false><<<dim3(DIM / 128, ROWS / 128), 256, 0, stream>>>(
      attnh, woutT, out, b_out, ROWS, DIM, INNER);
}
